// Matching_4105988735897
// MI455X (gfx1250) — matrix operand order repaired and checked by reading, not compiled, not run on hardware
//
#include <hip/hip_runtime.h>
#include <hip/hip_bf16.h>
#include <math.h>

typedef __attribute__((ext_vector_type(16))) __bf16 v16bf;
typedef __attribute__((ext_vector_type(8)))  float  v8f;

#define BB 4
#define SS 2048
#define TT 2048
#define CC 256

// src/16 . tgt/16 / 0.1  ==  raw_dot * (10/256)
__device__ __constant__ float SIM_SCALE = 0.0390625f;
#define CONF_THRESH 0.2f

// ---------------------------------------------------------------------------
// WMMA fragment helpers (wave32; contiguous k order k = 16*(lane/16) + element, same in A and B)
// ---------------------------------------------------------------------------

// A fragment 16x32 bf16 from a row-major bf16 matrix (ld elems per row)
__device__ inline v16bf fragA_bf16(const __bf16* base, int ld, int m0, int k0, int lane) {
    int row = m0 + (lane & 15);
    int kb  = k0 + ((lane & 16) ? 16 : 0);
    const __bf16* p = base + (size_t)row * ld + kb;
    v16bf a;
#pragma unroll
    for (int i = 0; i < 8; ++i) { a[i] = p[i]; a[i + 8] = p[i + 8]; }
    return a;
}

// A fragment 16x32 from a row-major f32 matrix, converting to bf16
__device__ inline v16bf fragA_f32(const float* base, int ld, int m0, int k0, int lane) {
    int row = m0 + (lane & 15);
    int kb  = k0 + ((lane & 16) ? 16 : 0);
    const float* p = base + (size_t)row * ld + kb;
    v16bf a;
#pragma unroll
    for (int i = 0; i < 8; ++i) { a[i] = (__bf16)p[i]; a[i + 8] = (__bf16)p[i + 8]; }
    return a;
}

// B fragment 32x16: column n of B == row n of the row-major "Bt" operand
// (we always compute A * Bt^T); lanes 0-15 hold K=k0..k0+15, lanes 16-31 K+16
__device__ inline v16bf fragB_bf16(const __bf16* baseT, int ld, int n0, int k0, int lane) {
    int col = n0 + (lane & 15);
    int kb  = k0 + ((lane & 16) ? 16 : 0);
    return *reinterpret_cast<const v16bf*>(baseT + (size_t)col * ld + kb);
}

__device__ inline v16bf fragB_f32(const float* baseT, int ld, int n0, int k0, int lane) {
    int col = n0 + (lane & 15);
    int kb  = k0 + ((lane & 16) ? 16 : 0);
    const float* p = baseT + (size_t)col * ld + kb;
    v16bf b;
#pragma unroll
    for (int i = 0; i < 16; ++i) b[i] = (__bf16)p[i];
    return b;
}

// ---------------------------------------------------------------------------
// Kernel 1: projection  P = X @ W^T  (X: [8192,256] f32, W: [256,256] f32)
// stores bf16; grid = (1024, 2): y==0 -> src, y==1 -> tgt; 8 waves/block,
// one 16x16 output tile per wave, K-loop of 8 WMMAs.
// ---------------------------------------------------------------------------
__global__ __launch_bounds__(256) void proj_kernel(
        const float* __restrict__ src, const float* __restrict__ tgt,
        const float* __restrict__ W,
        __bf16* __restrict__ Ps, __bf16* __restrict__ Pt) {
    const int lane = threadIdx.x & 31;
    const int wave = threadIdx.x >> 5;
    const float* X = blockIdx.y ? tgt : src;
    __bf16*      P = blockIdx.y ? Pt  : Ps;

    int tile = blockIdx.x * 8 + wave;       // 8192 tiles: 512 (M) x 16 (N)
    int m0 = (tile >> 4) << 4;
    int n0 = (tile & 15) << 4;

    v8f acc = {};
    for (int k = 0; k < CC; k += 32) {
        v16bf a = fragA_f32(X, CC, m0, k, lane);
        v16bf b = fragB_f32(W, CC, n0, k, lane);   // W row d, contiguous in c
        acc = __builtin_amdgcn_wmma_f32_16x16x32_bf16(
                  false, a, false, b, (short)0, acc, false, false);
    }
    int n  = n0 + (lane & 15);
    int mb = m0 + ((lane & 16) ? 8 : 0);
#pragma unroll
    for (int r = 0; r < 8; ++r)
        P[(size_t)(mb + r) * CC + n] = (__bf16)acc[r];
}

// ---------------------------------------------------------------------------
// Kernel 2: sim[b] = Ps[b] @ Pt[b]^T * SIM_SCALE    ([2048,2048] f32 per batch)
// Each wave computes a 16x64 strip (A-fragment reused across 4 WMMAs).
// grid = (512, B), 8 waves/block -> 4096 waves per batch.
// ---------------------------------------------------------------------------
__global__ __launch_bounds__(256) void sim_kernel(
        const __bf16* __restrict__ Ps, const __bf16* __restrict__ Pt,
        float* __restrict__ sim) {
    const int lane = threadIdx.x & 31;
    const int wave = threadIdx.x >> 5;
    const int b    = blockIdx.y;
    const __bf16* A  = Ps + (size_t)b * SS * CC;
    const __bf16* Bt = Pt + (size_t)b * TT * CC;
    float* out = sim + (size_t)b * SS * TT;

    int id = blockIdx.x * 8 + wave;         // 128 m-tiles x 32 n-strips
    int m0 = (id >> 5) << 4;
    int n0 = (id & 31) << 6;

    v8f acc0 = {}, acc1 = {}, acc2 = {}, acc3 = {};
    for (int k = 0; k < CC; k += 32) {
        v16bf a  = fragA_bf16(A, CC, m0, k, lane);
        v16bf b0 = fragB_bf16(Bt, CC, n0,      k, lane);
        v16bf b1 = fragB_bf16(Bt, CC, n0 + 16, k, lane);
        v16bf b2 = fragB_bf16(Bt, CC, n0 + 32, k, lane);
        v16bf b3 = fragB_bf16(Bt, CC, n0 + 48, k, lane);
        acc0 = __builtin_amdgcn_wmma_f32_16x16x32_bf16(false, a, false, b0, (short)0, acc0, false, false);
        acc1 = __builtin_amdgcn_wmma_f32_16x16x32_bf16(false, a, false, b1, (short)0, acc1, false, false);
        acc2 = __builtin_amdgcn_wmma_f32_16x16x32_bf16(false, a, false, b2, (short)0, acc2, false, false);
        acc3 = __builtin_amdgcn_wmma_f32_16x16x32_bf16(false, a, false, b3, (short)0, acc3, false, false);
    }
    const float sc = SIM_SCALE;
    int nn = lane & 15;
    int mb = m0 + ((lane & 16) ? 8 : 0);
#pragma unroll
    for (int r = 0; r < 8; ++r) {
        float* row = out + (size_t)(mb + r) * TT + n0 + nn;
        row[0]  = acc0[r] * sc;
        row[16] = acc1[r] * sc;
        row[32] = acc2[r] * sc;
        row[48] = acc3[r] * sc;
    }
}

// ---------------------------------------------------------------------------
// Kernel 3: column softmax stats (over valid s) per (b,t).  One thread per t,
// coalesced across the block; single online pass over S.
// ---------------------------------------------------------------------------
__global__ __launch_bounds__(256) void colstats_kernel(
        const float* __restrict__ sim, const unsigned char* __restrict__ smask,
        float* __restrict__ cmax, float* __restrict__ csum) {
    int g = blockIdx.x * 256 + threadIdx.x;    // over B*T
    int b = g >> 11;
    int t = g & (TT - 1);
    const float* col = sim + (size_t)b * SS * TT + t;
    const unsigned char* mk = smask + b * SS;
    float m = -__builtin_inff(), l = 0.f;
    for (int s = 0; s < SS; ++s) {
        if (mk[s]) {
            float x  = col[(size_t)s * TT];
            float nm = fmaxf(m, x);
            l = l * __expf(m - nm) + __expf(x - nm);
            m = nm;
        }
    }
    cmax[g] = m;
    csum[g] = l;
}

// ---------------------------------------------------------------------------
// Kernel 4: row softmax stats (over valid t) per (b,s).  One block per row,
// per-thread online accumulation + LDS tree combine.
// ---------------------------------------------------------------------------
__global__ __launch_bounds__(256) void rowstats_kernel(
        const float* __restrict__ sim, const unsigned char* __restrict__ tmask,
        float* __restrict__ rmax, float* __restrict__ rsum) {
    __shared__ float sm[256], sl[256];
    int b = blockIdx.y, s = blockIdx.x, tid = threadIdx.x;
    const float* row = sim + ((size_t)b * SS + s) * TT;
    const unsigned char* mk = tmask + b * TT;
    float m = -__builtin_inff(), l = 0.f;
    for (int t = tid; t < TT; t += 256) {
        if (mk[t]) {
            float x  = row[t];
            float nm = fmaxf(m, x);
            l = l * __expf(m - nm) + __expf(x - nm);
            m = nm;
        }
    }
    sm[tid] = m; sl[tid] = l;
    __syncthreads();
    for (int off = 128; off; off >>= 1) {
        if (tid < off) {
            float m2 = sm[tid + off], l2 = sl[tid + off];
            float M  = fmaxf(m, m2);
            float L  = ((l  > 0.f) ? l  * __expf(m  - M) : 0.f) +
                       ((l2 > 0.f) ? l2 * __expf(m2 - M) : 0.f);
            m = M; l = L; sm[tid] = m; sl[tid] = l;
        }
        __syncthreads();
    }
    if (tid == 0) { rmax[b * SS + s] = m; rsum[b * SS + s] = l; }
}

// ---------------------------------------------------------------------------
// Kernel 5: conf = softmax_col * softmax_row (0 where either mask invalid)
// ---------------------------------------------------------------------------
__global__ __launch_bounds__(256) void conf_kernel(
        const float* __restrict__ sim,
        const unsigned char* __restrict__ smask, const unsigned char* __restrict__ tmask,
        const float* __restrict__ cmax, const float* __restrict__ csum,
        const float* __restrict__ rmax, const float* __restrict__ rsum,
        float* __restrict__ conf) {
    size_t idx = (size_t)blockIdx.x * 256 + threadIdx.x;
    int b = (int)(idx >> 22);
    int rem = (int)(idx & ((1u << 22) - 1));
    int s = rem >> 11;
    int t = rem & (TT - 1);
    float v = 0.f;
    if (smask[b * SS + s] && tmask[b * TT + t]) {
        float x = sim[idx];
        v = __expf(x - cmax[b * TT + t]) * __expf(x - rmax[b * SS + s]) /
            (csum[b * TT + t] * rsum[b * SS + s]);
    }
    conf[idx] = v;
}

// ---------------------------------------------------------------------------
// Kernels 6/7: argmax values of conf per column (over s) and per row (over t)
// ---------------------------------------------------------------------------
__global__ __launch_bounds__(256) void conf_colmax_kernel(
        const float* __restrict__ conf, float* __restrict__ ccm) {
    int g = blockIdx.x * 256 + threadIdx.x;
    int b = g >> 11;
    int t = g & (TT - 1);
    const float* col = conf + (size_t)b * SS * TT + t;
    float m = 0.f;
    for (int s = 0; s < SS; ++s) m = fmaxf(m, col[(size_t)s * TT]);
    ccm[g] = m;
}

__global__ __launch_bounds__(256) void conf_rowmax_kernel(
        const float* __restrict__ conf, float* __restrict__ crm) {
    __shared__ float sm[256];
    int b = blockIdx.y, s = blockIdx.x, tid = threadIdx.x;
    const float* row = conf + ((size_t)b * SS + s) * TT;
    float m = 0.f;
    for (int t = tid; t < TT; t += 256) m = fmaxf(m, row[t]);
    sm[tid] = m;
    __syncthreads();
    for (int off = 128; off; off >>= 1) {
        if (tid < off) { m = fmaxf(m, sm[tid + off]); sm[tid] = m; }
        __syncthreads();
    }
    if (tid == 0) crm[b * SS + s] = m;
}

// ---------------------------------------------------------------------------
// Kernel 8: mutual-max + threshold mask -> 0/1 floats (overwrites sim region)
// ---------------------------------------------------------------------------
__global__ __launch_bounds__(256) void mask_kernel(
        const float* __restrict__ conf,
        const float* __restrict__ crm, const float* __restrict__ ccm,
        float* __restrict__ mout) {
    size_t idx = (size_t)blockIdx.x * 256 + threadIdx.x;
    int b = (int)(idx >> 22);
    int rem = (int)(idx & ((1u << 22) - 1));
    int s = rem >> 11;
    int t = rem & (TT - 1);
    float v = conf[idx];
    bool m = (v > CONF_THRESH) && (v == crm[b * SS + s]) && (v == ccm[b * TT + t]);
    mout[idx] = m ? 1.0f : 0.0f;
}

// ---------------------------------------------------------------------------
extern "C" void kernel_launch(void* const* d_in, const int* in_sizes, int n_in,
                              void* d_out, int out_size, void* d_ws, size_t ws_size,
                              hipStream_t stream) {
    const float*         src   = (const float*)d_in[0];
    const float*         tgt   = (const float*)d_in[1];
    const unsigned char* smask = (const unsigned char*)d_in[2];
    const unsigned char* tmask = (const unsigned char*)d_in[3];
    const float*         W     = (const float*)d_in[4];

    float* out  = (float*)d_out;
    float* conf = out;                         // first output [B,S,T]
    float* mout = out + (size_t)BB * SS * TT;  // second output [B,S,T]
    float* sim  = mout;                        // scratch: reuse mask region,
                                               // overwritten by mask_kernel last

    char* ws = (char*)d_ws;
    __bf16* Ps = (__bf16*)ws;                                  // 4 MiB
    __bf16* Pt = (__bf16*)(ws + (size_t)BB * SS * CC * 2);     // 4 MiB
    float* cmax = (float*)(ws + 2 * (size_t)BB * SS * CC * 2);
    float* csum = cmax + BB * TT;
    float* rmax = csum + BB * TT;
    float* rsum = rmax + BB * SS;
    float* ccm  = rsum + BB * SS;   // conf col-max per (b,t)
    float* crm  = ccm  + BB * TT;   // conf row-max per (b,s)

    proj_kernel<<<dim3(1024, 2), 256, 0, stream>>>(src, tgt, W, Ps, Pt);
    sim_kernel<<<dim3(512, BB), 256, 0, stream>>>(Ps, Pt, sim);
    colstats_kernel<<<(BB * TT) / 256, 256, 0, stream>>>(sim, smask, cmax, csum);
    rowstats_kernel<<<dim3(SS, BB), 256, 0, stream>>>(sim, tmask, rmax, rsum);
    conf_kernel<<<(BB * SS * TT) / 256, 256, 0, stream>>>(
        sim, smask, tmask, cmax, csum, rmax, rsum, conf);
    conf_colmax_kernel<<<(BB * TT) / 256, 256, 0, stream>>>(conf, ccm);
    conf_rowmax_kernel<<<dim3(SS, BB), 256, 0, stream>>>(conf, crm);
    mask_kernel<<<(BB * SS * TT) / 256, 256, 0, stream>>>(conf, crm, ccm, mout);
}
